// Compute2dAttentionLayer_23502061044077
// MI455X (gfx1250) — compile-verified, matrix operand order checked by reading, not run on hardware
//
#include <hip/hip_runtime.h>

// ---------------------------------------------------------------------------
// CDNA5 (gfx1250) 2D-attention decoder layer.
//   B=8, L=32, D=E=C=512, H=8, W=40, O=97
// Heavy GEMMs (3x3 conv as implicit GEMM, hs 1x1 conv) on
// v_wmma_f32_16x16x32_bf16, fed by double-buffered LDS tiles staged with
// global_load_async_to_lds_b128 (ASYNCcnt). Attention stage is a fused VALU
// kernel (tanh-dot -> softmax -> glimpse -> output projection).
// ---------------------------------------------------------------------------

#define USE_ASYNC_LDS 1

typedef __attribute__((ext_vector_type(16))) __bf16 v16bf;
typedef __attribute__((ext_vector_type(8)))  float  v8f;
typedef __attribute__((address_space(3))) unsigned int as3_uint;

union Frag { unsigned int u[8]; v16bf v; };

__device__ __forceinline__ unsigned short f32_to_bf16_rne(float f) {
    union { float f; unsigned int u; } cv; cv.f = f;
    unsigned int u = cv.u;
    u += 0x7FFFu + ((u >> 16) & 1u);   // round-to-nearest-even
    return (unsigned short)(u >> 16);
}

// 16-byte global -> LDS copy. Async (ASYNCcnt-tracked) when enabled.
__device__ __forceinline__ void cp16_g2l(unsigned int* l, const unsigned short* g) {
#if USE_ASYNC_LDS
    asm volatile("global_load_async_to_lds_b128 %0, %1, off"
                 :: "v"((unsigned)(unsigned long long)(as3_uint*)l),
                    "v"((unsigned long long)g)
                 : "memory");
#else
    const uint4 v = *(const uint4*)g;
    l[0] = v.x; l[1] = v.y; l[2] = v.z; l[3] = v.w;
#endif
}

__device__ __forceinline__ void stage_fence() {
#if USE_ASYNC_LDS
    asm volatile("s_wait_asynccnt 0" ::: "memory");
#endif
    __syncthreads();
}

// ---------------------------------------------------------------------------
// Stage 0 kernels: dtype conversion / layout preparation
// ---------------------------------------------------------------------------
__global__ void __launch_bounds__(256) cvt_f32_bf16_kernel(
    const float* __restrict__ src, unsigned short* __restrict__ dst, int n) {
    int i = blockIdx.x * 256 + threadIdx.x;
    if (i < n) dst[i] = f32_to_bf16_rne(src[i]);
}

// W[K,N] f32 -> WT[N,K] bf16 (coalesced writes)
__global__ void __launch_bounds__(256) transpose_bf16_kernel(
    const float* __restrict__ W, unsigned short* __restrict__ WT, int K, int N) {
    int i = blockIdx.x * 256 + threadIdx.x;
    if (i < K * N) {
        int n = i / K, k = i % K;
        WT[i] = f32_to_bf16_rne(W[(size_t)k * N + n]);
    }
}

// fmap [8,8,40,512] f32 -> zero-padded [8,10,42,512] bf16
__global__ void __launch_bounds__(256) pad_fmap_bf16_kernel(
    const float* __restrict__ fmap, unsigned short* __restrict__ pad) {
    int i = blockIdx.x * 256 + threadIdx.x;
    if (i >= 8 * 10 * 42 * 512) return;
    const int c = i & 511;
    const int p = i >> 9;            // (b*10 + h)*42 + w
    const int w = p % 42;
    const int h = (p / 42) % 10;
    const int b = p / (42 * 10);
    unsigned short v = 0;
    if (h >= 1 && h <= 8 && w >= 1 && w <= 40)
        v = f32_to_bf16_rne(fmap[((size_t)((b * 8 + h - 1) * 40) + (w - 1)) * 512 + c]);
    pad[i] = v;
}

// ---------------------------------------------------------------------------
// WMMA GEMM (TN):  C[M,N] = A[M,K](bf16) * BT[N,K](bf16)^T + bias[N]
// Tile 128(M) x 64(N), K-step 32, double-buffered LDS, async staging.
// 8 waves; wave w owns rows [w*16, w*16+16) x 64 N (4 accumulators).
// M%128==0, N%64==0, K%32==0 and K%8==0 alignment guaranteed by caller.
// ---------------------------------------------------------------------------
#define LROW 20   // LDS row stride in dwords (16 data + 4 pad, keeps 16B align)

__global__ void __launch_bounds__(256) gemm_bf16_tn_wmma_kernel(
    const unsigned short* __restrict__ A,   // [M,K]
    const unsigned short* __restrict__ BT,  // [N,K]
    const float* __restrict__ bias,         // [N]
    float* __restrict__ C,                  // [M,N]
    int M, int N, int K) {
    __shared__ unsigned int a_lds[2][128 * LROW];
    __shared__ unsigned int b_lds[2][64 * LROW];

    const int t    = threadIdx.x;
    const int lane = t & 31;
    const int wv   = t >> 5;
    const int mBase = blockIdx.y * 128;
    const int nBase = blockIdx.x * 64;

    const int ar = t >> 1, ah = t & 1;   // A stage: 2 x b128 per thread
    const int bn = t >> 2, bq = t & 3;   // B stage: 1 x b128 per thread
    const unsigned short* aSrc = A  + (size_t)(mBase + ar) * K + ah * 16;
    const unsigned short* bSrc = BT + (size_t)(nBase + bn) * K + bq * 8;

    auto stage = [&](int buf, int k0) {
        unsigned int* ad = &a_lds[buf][ar * LROW + ah * 8];
        cp16_g2l(ad,     aSrc + k0);
        cp16_g2l(ad + 4, aSrc + k0 + 8);
        cp16_g2l(&b_lds[buf][bn * LROW + bq * 4], bSrc + k0);
    };

    const v8f vzero = {0.f, 0.f, 0.f, 0.f, 0.f, 0.f, 0.f, 0.f};
    v8f acc[4] = {vzero, vzero, vzero, vzero};

    const int mrow = (wv << 4) + (lane & 15);
    const int koff = (lane >> 4) << 3;
    const int ncol = lane & 15;

    stage(0, 0);
    stage_fence();

    const int nk = K >> 5;
    for (int ks = 0; ks < nk; ++ks) {
        const int cur = ks & 1;
        if (ks + 1 < nk) stage(cur ^ 1, (ks + 1) << 5);

        Frag fa, fb0, fb1, fb2, fb3;
        const unsigned int* ap = &a_lds[cur][mrow * LROW + koff];
        const unsigned int* bp = &b_lds[cur][ncol * LROW + koff];
        #pragma unroll
        for (int i = 0; i < 8; ++i) fa.u[i]  = ap[i];
        #pragma unroll
        for (int i = 0; i < 8; ++i) fb0.u[i] = bp[i];
        #pragma unroll
        for (int i = 0; i < 8; ++i) fb1.u[i] = bp[16 * LROW + i];
        #pragma unroll
        for (int i = 0; i < 8; ++i) fb2.u[i] = bp[32 * LROW + i];
        #pragma unroll
        for (int i = 0; i < 8; ++i) fb3.u[i] = bp[48 * LROW + i];

        acc[0] = __builtin_amdgcn_wmma_f32_16x16x32_bf16(false, fa.v, false, fb0.v, (short)0, acc[0], false, false);
        acc[1] = __builtin_amdgcn_wmma_f32_16x16x32_bf16(false, fa.v, false, fb1.v, (short)0, acc[1], false, false);
        acc[2] = __builtin_amdgcn_wmma_f32_16x16x32_bf16(false, fa.v, false, fb2.v, (short)0, acc[2], false, false);
        acc[3] = __builtin_amdgcn_wmma_f32_16x16x32_bf16(false, fa.v, false, fb3.v, (short)0, acc[3], false, false);

        stage_fence();
    }

    // D layout: VGPR v -> M = v (lanes 0-15) / v+8 (lanes 16-31); N = lane&15
    const int mrow0 = mBase + (wv << 4) + ((lane >> 4) << 3);
    #pragma unroll
    for (int nb = 0; nb < 4; ++nb) {
        const int nc = nBase + (nb << 4) + (lane & 15);
        const float bi = bias[nc];
        #pragma unroll
        for (int v = 0; v < 8; ++v)
            C[(size_t)(mrow0 + v) * N + nc] = acc[nb][v] + bi;
    }
}

// ---------------------------------------------------------------------------
// 3x3 SAME conv as implicit GEMM over padded fmap: M=2560, N=512, K=9*512.
// Branch-free im2col staging (EXEC all-ones -> async + WMMA legal everywhere).
// ---------------------------------------------------------------------------
__global__ void __launch_bounds__(256) conv3x3_tn_wmma_kernel(
    const unsigned short* __restrict__ fpad,  // [8,10,42,512] bf16 zero-padded
    const unsigned short* __restrict__ wfT,   // [512(E), 4608(K)] bf16
    const float* __restrict__ bias,           // [512]
    float* __restrict__ fconv) {              // [2560,512]
    __shared__ unsigned int a_lds[2][128 * LROW];
    __shared__ unsigned int b_lds[2][64 * LROW];

    const int t    = threadIdx.x;
    const int lane = t & 31;
    const int wv   = t >> 5;
    const int mBase = blockIdx.y * 128;
    const int nBase = blockIdx.x * 64;

    const int ar = t >> 1, ah = t & 1;
    const int am   = mBase + ar;
    const int ab_  = am / 320;
    const int arem = am % 320;
    const int ay   = arem / 40, ax = arem % 40;
    // padded-space base for (kh,kw)=(0,0): (b, ay, ax)
    const unsigned short* aBase =
        fpad + ((size_t)((ab_ * 10 + ay) * 42 + ax)) * 512 + ah * 16;

    const int bn = t >> 2, bq = t & 3;
    const unsigned short* bSrc = wfT + (size_t)(nBase + bn) * 4608 + bq * 8;

    auto stage = [&](int buf, int ksn) {
        const int tap = ksn >> 4;            // 0..8
        const int cs  = (ksn & 15) << 5;     // channel slice
        const int kh  = tap / 3, kw = tap % 3;
        const unsigned short* as_ = aBase + (size_t)(kh * 42 + kw) * 512 + cs;
        unsigned int* ad = &a_lds[buf][ar * LROW + ah * 8];
        cp16_g2l(ad,     as_);
        cp16_g2l(ad + 4, as_ + 8);
        cp16_g2l(&b_lds[buf][bn * LROW + bq * 4], bSrc + ksn * 32);
    };

    const v8f vzero = {0.f, 0.f, 0.f, 0.f, 0.f, 0.f, 0.f, 0.f};
    v8f acc[4] = {vzero, vzero, vzero, vzero};

    const int mrow = (wv << 4) + (lane & 15);
    const int koff = (lane >> 4) << 3;
    const int ncol = lane & 15;

    stage(0, 0);
    stage_fence();

    const int nk = 144;                      // 9 taps * 512/32
    for (int ks = 0; ks < nk; ++ks) {
        const int cur = ks & 1;
        if (ks + 1 < nk) stage(cur ^ 1, ks + 1);

        Frag fa, fb0, fb1, fb2, fb3;
        const unsigned int* ap = &a_lds[cur][mrow * LROW + koff];
        const unsigned int* bp = &b_lds[cur][ncol * LROW + koff];
        #pragma unroll
        for (int i = 0; i < 8; ++i) fa.u[i]  = ap[i];
        #pragma unroll
        for (int i = 0; i < 8; ++i) fb0.u[i] = bp[i];
        #pragma unroll
        for (int i = 0; i < 8; ++i) fb1.u[i] = bp[16 * LROW + i];
        #pragma unroll
        for (int i = 0; i < 8; ++i) fb2.u[i] = bp[32 * LROW + i];
        #pragma unroll
        for (int i = 0; i < 8; ++i) fb3.u[i] = bp[48 * LROW + i];

        acc[0] = __builtin_amdgcn_wmma_f32_16x16x32_bf16(false, fa.v, false, fb0.v, (short)0, acc[0], false, false);
        acc[1] = __builtin_amdgcn_wmma_f32_16x16x32_bf16(false, fa.v, false, fb1.v, (short)0, acc[1], false, false);
        acc[2] = __builtin_amdgcn_wmma_f32_16x16x32_bf16(false, fa.v, false, fb2.v, (short)0, acc[2], false, false);
        acc[3] = __builtin_amdgcn_wmma_f32_16x16x32_bf16(false, fa.v, false, fb3.v, (short)0, acc[3], false, false);

        stage_fence();
    }

    const int mrow0 = mBase + (wv << 4) + ((lane >> 4) << 3);
    #pragma unroll
    for (int nb = 0; nb < 4; ++nb) {
        const int nc = nBase + (nb << 4) + (lane & 15);
        const float bi = bias[nc];
        #pragma unroll
        for (int v = 0; v < 8; ++v)
            fconv[(size_t)(mrow0 + v) * 512 + nc] = acc[nb][v] + bi;
    }
}

// ---------------------------------------------------------------------------
// Fused attention: one block per (b,l).
//   score[hw] = w_g . tanh(fconv[b,hw,:] + hs[b,l,:]) + b_g   (g never stored)
//   attn      = softmax(score);  glimpse = attn . fconv[b]
//   out[b,l]  = [glimpse, x[b,l]] @ w_out + b_out
// ---------------------------------------------------------------------------
__global__ void __launch_bounds__(256) attn_out_kernel(
    const float* __restrict__ x, const float* __restrict__ hs,
    const float* __restrict__ fconv, const float* __restrict__ w_g,
    const float* __restrict__ b_g, const float* __restrict__ w_out,
    const float* __restrict__ b_out, float* __restrict__ out) {
    __shared__ float sh_hs[512];
    __shared__ float sh_wg[512];
    __shared__ float sh_sc[320];
    __shared__ float sh_gl[512];
    __shared__ float red[8];

    const int t    = threadIdx.x;
    const int lane = t & 31;
    const int wv   = t >> 5;
    const int bl   = blockIdx.x;
    const int b    = bl >> 5;
    const float* frow = fconv + (size_t)b * 320 * 512;

    sh_hs[t]       = hs[(size_t)bl * 512 + t];
    sh_hs[t + 256] = hs[(size_t)bl * 512 + t + 256];
    sh_wg[t]       = w_g[t];
    sh_wg[t + 256] = w_g[t + 256];
    __syncthreads();

    for (int hw = wv; hw < 320; hw += 8) {
        const float* fp = frow + (size_t)hw * 512;
        float p = 0.f;
        #pragma unroll 4
        for (int j = 0; j < 16; ++j) {
            const int e = lane + (j << 5);
            p += sh_wg[e] * tanhf(fp[e] + sh_hs[e]);
        }
        for (int off = 16; off; off >>= 1) p += __shfl_xor(p, off, 32);
        if (lane == 0) sh_sc[hw] = p + b_g[0];
    }
    __syncthreads();

    float m = -3.4e38f;
    for (int i = t; i < 320; i += 256) m = fmaxf(m, sh_sc[i]);
    for (int off = 16; off; off >>= 1) m = fmaxf(m, __shfl_xor(m, off, 32));
    if (lane == 0) red[wv] = m;
    __syncthreads();
    if (t == 0) {
        float mm = red[0];
        for (int i = 1; i < 8; ++i) mm = fmaxf(mm, red[i]);
        red[0] = mm;
    }
    __syncthreads();
    const float gmax = red[0];
    __syncthreads();

    float s = 0.f;
    for (int i = t; i < 320; i += 256) {
        const float e = expf(sh_sc[i] - gmax);
        sh_sc[i] = e;
        s += e;
    }
    for (int off = 16; off; off >>= 1) s += __shfl_xor(s, off, 32);
    if (lane == 0) red[wv] = s;
    __syncthreads();
    if (t == 0) {
        float ss = 0.f;
        for (int i = 0; i < 8; ++i) ss += red[i];
        red[0] = ss;
    }
    __syncthreads();
    const float inv = 1.f / red[0];
    __syncthreads();

    for (int e = t; e < 512; e += 256) {
        float g = 0.f;
        const float* fp = frow + e;
        for (int hw = 0; hw < 320; ++hw) g += sh_sc[hw] * fp[(size_t)hw * 512];
        sh_gl[e] = g * inv;
    }
    __syncthreads();

    if (t < 97) {
        float acc = b_out[t];
        for (int c = 0; c < 512; ++c) acc += sh_gl[c] * w_out[(size_t)c * 97 + t];
        const float* xr = x + (size_t)bl * 512;
        for (int c = 0; c < 512; ++c) acc += xr[c] * w_out[(size_t)(512 + c) * 97 + t];
        out[(size_t)bl * 97 + t] = acc;
    }
}

// ---------------------------------------------------------------------------
// Host launcher
// ---------------------------------------------------------------------------
extern "C" void kernel_launch(void* const* d_in, const int* in_sizes, int n_in,
                              void* d_out, int out_size, void* d_ws, size_t ws_size,
                              hipStream_t stream) {
    (void)in_sizes; (void)n_in; (void)out_size; (void)ws_size;

    const float* x      = (const float*)d_in[0];  // [8,32,512]
    const float* fmap   = (const float*)d_in[1];  // [8,8,40,512]
    const float* w_hs   = (const float*)d_in[2];  // [512,512]
    const float* b_hs   = (const float*)d_in[3];  // [512]
    const float* w_fmap = (const float*)d_in[4];  // [3,3,512,512]
    const float* b_fmap = (const float*)d_in[5];  // [512]
    const float* w_g    = (const float*)d_in[6];  // [512]
    const float* b_g    = (const float*)d_in[7];  // [1]
    const float* w_out  = (const float*)d_in[8];  // [1024,97]
    const float* b_out  = (const float*)d_in[9];  // [97]
    float* out = (float*)d_out;                   // [1,256,97]

    const int N_X   = 256 * 512;        // 131072
    const int N_WH  = 512 * 512;        // 262144
    const int N_PAD = 8 * 10 * 42 * 512;// 1720320
    const int N_WF  = 9 * 512 * 512;    // 2359296
    const int N_FC  = 2560 * 512;       // 1310720

    char* ws = (char*)d_ws;
    size_t off = 0;
    auto take = [&](size_t bytes) {
        size_t r = off;
        off += (bytes + 255) & ~(size_t)255;
        return r;
    };
    unsigned short* x_bf  = (unsigned short*)(ws + take((size_t)N_X   * 2));
    unsigned short* whT   = (unsigned short*)(ws + take((size_t)N_WH  * 2)); // [N=512,K=512]
    unsigned short* fpad  = (unsigned short*)(ws + take((size_t)N_PAD * 2));
    unsigned short* wfT   = (unsigned short*)(ws + take((size_t)N_WF  * 2)); // [N=512,K=4608]
    float* hs    = (float*)(ws + take((size_t)N_X  * 4));   // [256,512]
    float* fconv = (float*)(ws + take((size_t)N_FC * 4));   // [2560,512]

    // Stage 0: conversions / layout prep
    cvt_f32_bf16_kernel<<<(N_X + 255) / 256, 256, 0, stream>>>(x, x_bf, N_X);
    transpose_bf16_kernel<<<(N_WH + 255) / 256, 256, 0, stream>>>(w_hs, whT, 512, 512);
    pad_fmap_bf16_kernel<<<(N_PAD + 255) / 256, 256, 0, stream>>>(fmap, fpad);
    transpose_bf16_kernel<<<(N_WF + 255) / 256, 256, 0, stream>>>(w_fmap, wfT, 4608, 512);

    // Stage 1: hs = x @ w_hs + b_hs   (M=256, N=512, K=512)
    gemm_bf16_tn_wmma_kernel<<<dim3(512 / 64, 256 / 128), 256, 0, stream>>>(
        x_bf, whT, b_hs, hs, 256, 512, 512);

    // Stage 2: fconv = conv3x3(fmap, w_fmap) + b_fmap  (implicit GEMM)
    conv3x3_tn_wmma_kernel<<<dim3(512 / 64, 2560 / 128), 256, 0, stream>>>(
        fpad, wfT, b_fmap, fconv);

    // Stage 3: fused attention + softmax + glimpse + output projection
    attn_out_kernel<<<256, 256, 0, stream>>>(
        x, hs, fconv, w_g, b_g, w_out, b_out, out);
}
